// GIFStreamEvaluator_24352464570195
// MI455X (gfx1250) — matrix operand order repaired and checked by reading, not compiled, not run on hardware
//
#include <hip/hip_runtime.h>
#include <math.h>

typedef __attribute__((ext_vector_type(16))) __bf16 v16bf;
typedef __attribute__((ext_vector_type(8)))  float  v8f;
typedef unsigned int  u32;
typedef unsigned short u16;

// ---- problem constants ----
#define MROWS   300000
#define NTILES  18750      // MROWS / 16
#define NPAIRS  9375       // NTILES / 2 (M-block = 2 tiles per wave iteration)
#define OUTS    160
#define WAVES   4          // waves per block (128 threads)
#define GOPC    30

// feat layout (K index, remapped so everything 2-aligned):
//   0..31 af | 32..47 tfeat | 48..50 vd | 51..63 pad
// feat_ layout: 0..31 af (shared chunk0!) | 32..47 tfeat | 48..55 emb | 56..63 pad
// Only cols 32..63 of feat_ are stored (half buffer), chunk0 A-regs are reused.

// ---- swizzled weight blocks (each block = 32 lanes x 16 bf16 = 512 u16) ----
#define BLK_W1   0    // 48 : shared layer1 (o|c|v) 24 Ntiles x 2 Kchunks (rows remapped)
#define BLK_MW1  48   // 16 : motion layer1 8 Ntiles x 2 Kchunks
#define BLK_W2   64   // 32 : layer2 (o:1,c:2,v:5 Ntiles) x 4 Kchunks
#define BLK_MW2  96   // 12 : motion layer2 3 Ntiles x 4 Kchunks
#define NBLK     108

union BfVec {
  v16bf v;
  u32   u[8];
  uint4 q[2];
};

// RNE conversion (used once for weights)
__device__ __forceinline__ u16 f2bf(float f) {
  u32 x = __float_as_uint(f);
  u32 r = x + 0x7FFFu + ((x >> 16) & 1u);
  return (u16)(r >> 16);
}
// fast round-half-up conversion (activations): 1 VALU, store folds to d16_hi
__device__ __forceinline__ u16 f2bf_fast(float f) {
  return (u16)((__float_as_uint(f) + 0x8000u) >> 16);
}
// fused relu + bf16: bit-domain relu (v_max_i32) + round (v_add) -> 2 VALU, d16_hi store
__device__ __forceinline__ u16 f2bf_relu(float f) {
  int b = (int)__float_as_uint(f);
  int m = (b > 0) ? b : 0;                 // v_max_i32: relu for non-NaN floats
  return (u16)(((u32)m + 0x8000u) >> 16);
}
// packed pair via v_perm_b32: 2 adds + 1 perm
__device__ __forceinline__ u32 pk2bf(float a, float b) {
  u32 x = __float_as_uint(a) + 0x8000u;
  u32 y = __float_as_uint(b) + 0x8000u;
  return __builtin_amdgcn_perm(y, x, 0x07060302u);   // {y.hi16, x.hi16}
}
__device__ __forceinline__ float sigmoidf_(float x) { return 1.0f / (1.0f + __expf(-x)); }

__device__ __forceinline__ v8f wmma_bf16(const BfVec& a, const BfVec& b, v8f c) {
  return __builtin_amdgcn_wmma_f32_16x16x32_bf16(false, a.v, false, b.v, (short)0, c, false, false);
}
__device__ __forceinline__ BfVec loadB(const u16* w, int blk, int lane) {
  BfVec b;
  const uint4* p = (const uint4*)&w[(blk * 32 + lane) * 16];
  b.q[0] = p[0]; b.q[1] = p[1];
  return b;
}

__global__ __launch_bounds__(128)
void gifstream_kernel(const float* __restrict__ af,     const float* __restrict__ anchors,
                      const float* __restrict__ scales, const float* __restrict__ timef,
                      const float* __restrict__ factors,const float* __restrict__ c2w,
                      const float* __restrict__ tvalp,
                      const float* __restrict__ ow1, const float* __restrict__ ob1,
                      const float* __restrict__ ow2, const float* __restrict__ ob2,
                      const float* __restrict__ cw1, const float* __restrict__ cb1,
                      const float* __restrict__ cw2, const float* __restrict__ cb2,
                      const float* __restrict__ vw1, const float* __restrict__ vb1,
                      const float* __restrict__ vw2, const float* __restrict__ vb2,
                      const float* __restrict__ mw1, const float* __restrict__ mb1,
                      const float* __restrict__ mw2, const float* __restrict__ mb2,
                      const int*   __restrict__ vis,
                      float* __restrict__ out)
{
  __shared__ __align__(16) u16  ldsW[NBLK * 512];      // 108 KB swizzled bf16 weights
  __shared__ float ldsB1[384];
  __shared__ float ldsMB1[128];
  __shared__ float ldsB2[128];
  __shared__ float ldsMB2[48];
  // per-wave, per-subtile unified buffer (4 KB = 2048 u16):
  //   staging phase: [0..1024) feat 16x64 ; [1024..1536) feat_ cols32-63 16x32
  //   compute phase: [0..2048) hidden 16x128 (overlays)
  __shared__ __align__(16) u16  ldsT[WAVES][2][2048];
  __shared__ float ldsFA[WAVES][2][32];                // pf, mf per row

  const int tid = threadIdx.x;

  // ---------------- stage weights into LDS (B-matrix register image) ----------------
  // B layout (16-bit, 32x16): lane L -> N = L&15 ; element i -> K = (i>=8?16:0) + 8*(L>>4) + (i&7)
  for (int e = tid; e < NBLK * 512; e += blockDim.x) {
    int blk = e >> 9, within = e & 511;
    int L = within >> 4, i = within & 15;
    int n  = L & 15;
    int kk = ((i & 8) << 1) + ((L >> 4) << 3) + (i & 7);
    float val = 0.0f;
    if (blk < BLK_MW1) {                       // shared layer-1, rows remapped to feat layout
      int t = blk >> 1, c = blk & 1;
      int K = c * 32 + kk, Nc = t * 16 + n;
      int sK = (K < 32) ? K : (K < 48 ? K + 3 : (K < 51 ? K - 16 : -1));
      if (sK >= 0) {
        const float* w = (Nc < 128) ? ow1 : (Nc < 256 ? cw1 : vw1);
        val = w[sK * 128 + (Nc & 127)];
      }
    } else if (blk < BLK_W2) {                 // motion layer-1 (56x128, natural order)
      int b = blk - BLK_MW1, t = b >> 1, c = b & 1;
      int K = c * 32 + kk;
      if (K < 56) val = mw1[K * 128 + t * 16 + n];
    } else if (blk < BLK_MW2) {                // layer-2 cat, per-MLP padded tiles
      int b = blk - BLK_W2, t = b >> 2, c = b & 3;
      int K = c * 32 + kk;
      if (t == 0)      { if (n < 10) val = ow2[K * 10 + n]; }
      else if (t < 3)  { int j = (t - 1) * 16 + n; if (j < 30) val = cw2[K * 30 + j]; }
      else             { int j = (t - 3) * 16 + n; if (j < 70) val = vw2[K * 70 + j]; }
    } else {                                   // motion layer-2 (128x37 -> 48)
      int b = blk - BLK_MW2, t = b >> 2, c = b & 3;
      int K = c * 32 + kk;
      int j = t * 16 + n;
      if (j < 37) val = mw2[K * 37 + j];
    }
    ldsW[e] = f2bf(val);
  }
  for (int j = tid; j < 384; j += blockDim.x)
    ldsB1[j] = (j < 128) ? ob1[j] : (j < 256 ? cb1[j - 128] : vb1[j - 256]);
  for (int j = tid; j < 128; j += blockDim.x) ldsMB1[j] = mb1[j];
  for (int j = tid; j < 128; j += blockDim.x) {
    int t = j >> 4, n = j & 15; float v = 0.0f;
    if (t == 0)      { if (n < 10) v = ob2[n]; }
    else if (t < 3)  { int q = (t - 1) * 16 + n; if (q < 30) v = cb2[q]; }
    else             { int q = (t - 3) * 16 + n; if (q < 70) v = vb2[q]; }
    ldsB2[j] = v;
  }
  for (int j = tid; j < 48; j += blockDim.x) ldsMB2[j] = (j < 37) ? mb2[j] : 0.0f;
  __syncthreads();

  // ---------------- time / camera scalars ----------------
  const float tval = tvalp[0];
  const int   feat_start = (int)floorf(tval * (float)(GOPC - 1));
  const float camx = c2w[3], camy = c2w[7], camz = c2w[11];
  float emb[8];
  {
    float fr = 3.14159265358979323846f * tval;
    #pragma unroll
    for (int h = 0; h < 4; ++h) { emb[h] = sinf(fr); emb[4 + h] = cosf(fr); fr *= 2.0f; }
  }

  const int wv    = tid >> 5;
  const int lane  = tid & 31;
  const int mrow  = lane & 15;
  const int hi    = lane >> 4;
  const int hi8   = hi * 8;
  const int nlane = lane & 15;

  for (int tp = blockIdx.x * WAVES + wv; tp < NPAIRS; tp += gridDim.x * WAVES) {
    // ---- gather & stage both subtiles ----
    #pragma unroll
    for (int s = 0; s < 2; ++s) {
      const int row = tp * 32 + s * 16 + mrow;
      const int idx = vis[row];
      u16* T = &ldsT[wv][s][0];
      {
        const float4* src4 = (const float4*)(af + (size_t)idx * 32 + hi * 16);
        const int base = mrow * 64 + hi * 16;
        #pragma unroll
        for (int q = 0; q < 4; ++q) {
          float4 f = src4[q];
          *(u32*)&T[base + q * 4 + 0] = pk2bf(f.x, f.y);
          *(u32*)&T[base + q * 4 + 2] = pk2bf(f.z, f.w);
        }
      }
      if (hi == 0) {
        float fa0 = sigmoidf_(factors[(size_t)idx * 4 + 0]);
        float fa1 = sigmoidf_(factors[(size_t)idx * 4 + 1]);
        float fa2 = sigmoidf_(factors[(size_t)idx * 4 + 2]);
        float fa3 = sigmoidf_(factors[(size_t)idx * 4 + 3]);
        float pf = fa3, mf = fa1, tff = fa0;
        ldsFA[wv][s][mrow * 2 + 0] = pf;
        ldsFA[wv][s][mrow * 2 + 1] = mf;

        // tfeat: feat cols 32..47 and feat_ half-buffer cols 0..15
        const float4* t4 = (const float4*)(timef + ((size_t)idx * GOPC + feat_start) * 16);
        #pragma unroll
        for (int q = 0; q < 4; ++q) {
          float4 f = t4[q];
          u32 p0 = pk2bf(f.x * tff, f.y * tff);
          u32 p1 = pk2bf(f.z * tff, f.w * tff);
          *(u32*)&T[mrow * 64 + 32 + q * 4 + 0] = p0;
          *(u32*)&T[mrow * 64 + 32 + q * 4 + 2] = p1;
          *(u32*)&T[1024 + mrow * 32 + q * 4 + 0] = p0;
          *(u32*)&T[1024 + mrow * 32 + q * 4 + 2] = p1;
        }
        // vd: feat cols 48..50, pad 51..63
        float ax = anchors[(size_t)idx * 3 + 0];
        float ay = anchors[(size_t)idx * 3 + 1];
        float az = anchors[(size_t)idx * 3 + 2];
        float dx = ax - camx, dy = ay - camy, dz = az - camz;
        float inv = 1.0f / fmaxf(sqrtf(dx * dx + dy * dy + dz * dz), 1e-8f);
        *(u32*)&T[mrow * 64 + 48] = pk2bf(dx * inv, dy * inv);
        *(u32*)&T[mrow * 64 + 50] = (u32)f2bf_fast(dz * inv);   // col 50 + zero pad col 51
        #pragma unroll
        for (int q = 0; q < 6; ++q) *(u32*)&T[mrow * 64 + 52 + q * 2] = 0;
        // feat_ half: emb (local 16..23), pad (local 24..31)
        #pragma unroll
        for (int hq = 0; hq < 4; ++hq)
          *(u32*)&T[1024 + mrow * 32 + 16 + hq * 2] = pk2bf(emb[2 * hq], emb[2 * hq + 1]);
        #pragma unroll
        for (int q = 0; q < 4; ++q) *(u32*)&T[1024 + mrow * 32 + 24 + q * 2] = 0;

        // direct outputs: sc(6) fa(4) an(3) at cols 147..159
        float* ob = out + (size_t)row * OUTS;
        ob[147] = __expf(scales[(size_t)idx * 6 + 0]);
        ob[148] = __expf(scales[(size_t)idx * 6 + 1]);
        ob[149] = __expf(scales[(size_t)idx * 6 + 2]);
        ob[150] = __expf(scales[(size_t)idx * 6 + 3]) * pf;
        ob[151] = __expf(scales[(size_t)idx * 6 + 4]) * pf;
        ob[152] = __expf(scales[(size_t)idx * 6 + 5]) * pf;
        ob[153] = fa0; ob[154] = fa1; ob[155] = fa2; ob[156] = fa3;
        ob[157] = ax;  ob[158] = ay;  ob[159] = az;
      }
    }
    asm volatile("s_wait_dscnt 0" ::: "memory");

    // ---- A register images (16-bit A layout: K = (v>=4?16:0) + 8*hi + 2*(v&3) + h) ----
    BfVec aC0[2], aC1[2], amC1[2];     // chunk0 shared between feat and feat_
    #pragma unroll
    for (int s = 0; s < 2; ++s) {
      u16* T = &ldsT[wv][s][0];
      #pragma unroll
      for (int v = 0; v < 8; ++v) {
        int kk = ((v & 4) << 2) + hi8 + ((v & 3) << 1);
        aC0[s].u[v]  = *(const u32*)&T[mrow * 64 + kk];
        aC1[s].u[v]  = *(const u32*)&T[mrow * 64 + 32 + kk];
        amC1[s].u[v] = *(const u32*)&T[1024 + mrow * 32 + kk];
      }
    }
    float pfv[2][8], mfv[2][8];
    #pragma unroll
    for (int s = 0; s < 2; ++s)
      #pragma unroll
      for (int r = 0; r < 8; ++r) {
        pfv[s][r] = ldsFA[wv][s][(r + hi8) * 2 + 0];
        mfv[s][r] = ldsFA[wv][s][(r + hi8) * 2 + 1];
      }

    // ---- three shared-input MLPs (opac / cols / cov) ----
    for (int mlp = 0; mlp < 3; ++mlp) {
      for (int tl = 0; tl < 8; ++tl) {                 // layer-1 hidden tiles
        int t = mlp * 8 + tl;
        float bias = ldsB1[t * 16 + nlane];
        v8f acc0, acc1;
        #pragma unroll
        for (int r = 0; r < 8; ++r) { acc0[r] = bias; acc1[r] = bias; }
        BfVec b0 = loadB(ldsW, BLK_W1 + t * 2 + 0, lane);
        BfVec b1 = loadB(ldsW, BLK_W1 + t * 2 + 1, lane);
        acc0 = wmma_bf16(aC0[0], b0, acc0);
        acc1 = wmma_bf16(aC0[1], b0, acc1);
        acc0 = wmma_bf16(aC1[0], b1, acc0);
        acc1 = wmma_bf16(aC1[1], b1, acc1);
        u16* h0 = &ldsT[wv][0][0];
        u16* h1 = &ldsT[wv][1][0];
        #pragma unroll
        for (int r = 0; r < 8; ++r) {
          h0[(r + hi8) * 128 + tl * 16 + nlane] = f2bf_relu(acc0[r]);
          h1[(r + hi8) * 128 + tl * 16 + nlane] = f2bf_relu(acc1[r]);
        }
      }
      asm volatile("s_wait_dscnt 0" ::: "memory");
      BfVec ha[2][4];
      #pragma unroll
      for (int s = 0; s < 2; ++s) {
        u16* T = &ldsT[wv][s][0];
        #pragma unroll
        for (int c = 0; c < 4; ++c)
          #pragma unroll
          for (int v = 0; v < 8; ++v) {
            int kk = ((v & 4) << 2) + hi8 + ((v & 3) << 1);
            ha[s][c].u[v] = *(const u32*)&T[mrow * 128 + c * 32 + kk];
          }
      }
      const int t0    = (mlp == 0) ? 0 : (mlp == 1 ? 1 : 3);
      const int ntl   = (mlp == 0) ? 1 : (mlp == 1 ? 2 : 5);
      const int width = (mlp == 0) ? 10 : (mlp == 1 ? 30 : 70);
      const int cofs  = (mlp == 0) ? 0 : (mlp == 1 ? 10 : 40);
      for (int tt = 0; tt < ntl; ++tt) {               // layer-2 output tiles
        int t = t0 + tt;
        float bias = ldsB2[t * 16 + nlane];
        v8f acc0, acc1;
        #pragma unroll
        for (int r = 0; r < 8; ++r) { acc0[r] = bias; acc1[r] = bias; }
        #pragma unroll
        for (int c = 0; c < 4; ++c) {
          BfVec b = loadB(ldsW, BLK_W2 + t * 4 + c, lane);
          acc0 = wmma_bf16(ha[0][c], b, acc0);
          acc1 = wmma_bf16(ha[1][c], b, acc1);
        }
        int j = tt * 16 + nlane;
        if (j < width) {
          #pragma unroll
          for (int s = 0; s < 2; ++s) {
            const v8f& acc = s ? acc1 : acc0;
            #pragma unroll
            for (int r = 0; r < 8; ++r) {
              float vo = acc[r];
              if (mlp == 0)      vo = tanhf(vo) * pfv[s][r];
              else if (mlp == 1) vo = sigmoidf_(vo);
              out[(size_t)(tp * 32 + s * 16 + r + hi8) * OUTS + cofs + j] = vo;
            }
          }
        }
      }
    }

    // ---- motion MLP (feat_ = chunk0 shared + amC1) ----
    for (int tl = 0; tl < 8; ++tl) {
      float bias = ldsMB1[tl * 16 + nlane];
      v8f acc0, acc1;
      #pragma unroll
      for (int r = 0; r < 8; ++r) { acc0[r] = bias; acc1[r] = bias; }
      BfVec b0 = loadB(ldsW, BLK_MW1 + tl * 2 + 0, lane);
      BfVec b1 = loadB(ldsW, BLK_MW1 + tl * 2 + 1, lane);
      acc0 = wmma_bf16(aC0[0],  b0, acc0);
      acc1 = wmma_bf16(aC0[1],  b0, acc1);
      acc0 = wmma_bf16(amC1[0], b1, acc0);
      acc1 = wmma_bf16(amC1[1], b1, acc1);
      u16* h0 = &ldsT[wv][0][0];
      u16* h1 = &ldsT[wv][1][0];
      #pragma unroll
      for (int r = 0; r < 8; ++r) {
        h0[(r + hi8) * 128 + tl * 16 + nlane] = f2bf_relu(acc0[r]);
        h1[(r + hi8) * 128 + tl * 16 + nlane] = f2bf_relu(acc1[r]);
      }
    }
    asm volatile("s_wait_dscnt 0" ::: "memory");
    {
      BfVec ha[2][4];
      #pragma unroll
      for (int s = 0; s < 2; ++s) {
        u16* T = &ldsT[wv][s][0];
        #pragma unroll
        for (int c = 0; c < 4; ++c)
          #pragma unroll
          for (int v = 0; v < 8; ++v) {
            int kk = ((v & 4) << 2) + hi8 + ((v & 3) << 1);
            ha[s][c].u[v] = *(const u32*)&T[mrow * 128 + c * 32 + kk];
          }
      }
      for (int tt = 0; tt < 3; ++tt) {
        float bias = ldsMB2[tt * 16 + nlane];
        v8f acc0, acc1;
        #pragma unroll
        for (int r = 0; r < 8; ++r) { acc0[r] = bias; acc1[r] = bias; }
        #pragma unroll
        for (int c = 0; c < 4; ++c) {
          BfVec b = loadB(ldsW, BLK_MW2 + tt * 4 + c, lane);
          acc0 = wmma_bf16(ha[0][c], b, acc0);
          acc1 = wmma_bf16(ha[1][c], b, acc1);
        }
        int j = tt * 16 + nlane;
        if (j < 37) {
          #pragma unroll
          for (int s = 0; s < 2; ++s) {
            const v8f& acc = s ? acc1 : acc0;
            #pragma unroll
            for (int r = 0; r < 8; ++r)
              out[(size_t)(tp * 32 + s * 16 + r + hi8) * OUTS + 110 + j] = acc[r] * mfv[s][r];
          }
        }
      }
    }
  }
}

extern "C" void kernel_launch(void* const* d_in, const int* in_sizes, int n_in,
                              void* d_out, int out_size, void* d_ws, size_t ws_size,
                              hipStream_t stream) {
  (void)in_sizes; (void)n_in; (void)out_size; (void)d_ws; (void)ws_size;
  const float* af      = (const float*)d_in[0];
  const float* anchors = (const float*)d_in[1];
  const float* scales  = (const float*)d_in[2];
  const float* timef   = (const float*)d_in[3];
  const float* factors = (const float*)d_in[4];
  const float* c2w     = (const float*)d_in[5];
  const float* tvalp   = (const float*)d_in[6];
  const float* ow1 = (const float*)d_in[7];
  const float* ob1 = (const float*)d_in[8];
  const float* ow2 = (const float*)d_in[9];
  const float* ob2 = (const float*)d_in[10];
  const float* cw1 = (const float*)d_in[11];
  const float* cb1 = (const float*)d_in[12];
  const float* cw2 = (const float*)d_in[13];
  const float* cb2 = (const float*)d_in[14];
  const float* vw1 = (const float*)d_in[15];
  const float* vb1 = (const float*)d_in[16];
  const float* vw2 = (const float*)d_in[17];
  const float* vb2 = (const float*)d_in[18];
  const float* mw1 = (const float*)d_in[19];
  const float* mb1 = (const float*)d_in[20];
  const float* mw2 = (const float*)d_in[21];
  const float* mb2 = (const float*)d_in[22];
  const int*   vis = (const int*)d_in[23];
  float* out = (float*)d_out;

  dim3 grid((NPAIRS + WAVES - 1) / WAVES);
  dim3 block(128);
  hipLaunchKernelGGL(gifstream_kernel, grid, block, 0, stream,
                     af, anchors, scales, timef, factors, c2w, tvalp,
                     ow1, ob1, ow2, ob2, cw1, cb1, cw2, cb2,
                     vw1, vb1, vw2, vb2, mw1, mb1, mw2, mb2,
                     vis, out);
}
